// NeuralOdeFunction_88579405513274
// MI455X (gfx1250) — matrix operand order repaired and checked by reading, not compiled, not run on hardware
//
#include <hip/hip_runtime.h>
#include <hip/hip_bf16.h>

// ---------------------------------------------------------------------------
// CDNA5 (gfx1250) transformer forward using v_wmma_f32_16x16x32_bf16.
// wave32; one wave per 16-row output tile; LDS staging between fused stages.
// qkv / attention-out intermediates are bf16-resident: converted once at
// store, loaded directly as WMMA fragments (no per-load cvt in the hot path).
// ---------------------------------------------------------------------------

typedef __attribute__((ext_vector_type(16))) __bf16 v16bf;
typedef __attribute__((ext_vector_type(8)))  __bf16 v8bf;
typedef __attribute__((ext_vector_type(8)))  float  v8f;

#define BATCH 16
#define SEQ   1024
#define HID   64
#define MLPH  128
#define FF    256
#define NHEAD 4
#define HD    16
#define MAXF  512
#define NTOK  256

// Branchless tanh: saturates to +/-1 via __expf overflow/underflow.
__device__ __forceinline__ float fast_tanh(float v) {
    return 1.f - 2.f / (__expf(2.f * v) + 1.f);
}

// ---- fragment loaders (ISA 7.12.2 layouts), f32 sources -------------------
// A (16x32 bf16): lane m = lane&15, half kh = 0/8; elements j=0..7 -> K=kh+0..7,
// j=8..15 -> K=16+kh..16+kh+7  => two runs of 8 consecutive values.
__device__ __forceinline__ v16bf load_a32(const float* p, int lda, int row0,
                                          int k0, int lane) {
    int m  = lane & 15;
    int kh = (lane >> 4) << 3;
    const float4* r4 = (const float4*)(p + (size_t)(row0 + m) * lda + k0 + kh);
    float4 x0 = r4[0], x1 = r4[1], x2 = r4[4], x3 = r4[5];
    v16bf a;
    a[0]  = (__bf16)x0.x; a[1]  = (__bf16)x0.y; a[2]  = (__bf16)x0.z; a[3]  = (__bf16)x0.w;
    a[4]  = (__bf16)x1.x; a[5]  = (__bf16)x1.y; a[6]  = (__bf16)x1.z; a[7]  = (__bf16)x1.w;
    a[8]  = (__bf16)x2.x; a[9]  = (__bf16)x2.y; a[10] = (__bf16)x2.z; a[11] = (__bf16)x2.w;
    a[12] = (__bf16)x3.x; a[13] = (__bf16)x3.y; a[14] = (__bf16)x3.z; a[15] = (__bf16)x3.w;
    return a;
}

// B^T (32x16): lane n = lane&15, half kh = 0/8; elements j=0..7 -> K=kh+0..7,
// j=8..15 -> K=16+kh..16+kh+7.  B[k][n] = W[(n0+n)*ldw + k0+k].
__device__ __forceinline__ v16bf load_bt32(const float* p, int ldw, int n0,
                                           int k0, int lane) {
    int n  = lane & 15;
    int kh = (lane >> 4) << 3;
    const float4* r4 = (const float4*)(p + (size_t)(n0 + n) * ldw + k0 + kh);
    float4 x0 = r4[0], x1 = r4[1], x2 = r4[4], x3 = r4[5];
    v16bf b;
    b[0]  = (__bf16)x0.x; b[1]  = (__bf16)x0.y; b[2]  = (__bf16)x0.z; b[3]  = (__bf16)x0.w;
    b[4]  = (__bf16)x1.x; b[5]  = (__bf16)x1.y; b[6]  = (__bf16)x1.z; b[7]  = (__bf16)x1.w;
    b[8]  = (__bf16)x2.x; b[9]  = (__bf16)x2.y; b[10] = (__bf16)x2.z; b[11] = (__bf16)x2.w;
    b[12] = (__bf16)x3.x; b[13] = (__bf16)x3.y; b[14] = (__bf16)x3.z; b[15] = (__bf16)x3.w;
    return b;
}

// ---- fragment loaders, bf16 sources (no conversion) -----------------------
__device__ __forceinline__ v16bf load_a32_bf(const __bf16* p, int lda, int row0,
                                             int k0, int lane) {
    int m  = lane & 15;
    int kh = (lane >> 4) << 3;
    const __bf16* r = p + (size_t)(row0 + m) * lda + k0 + kh;
    v8bf lo = *(const v8bf*)(r);
    v8bf hi = *(const v8bf*)(r + 16);
    v16bf a;
#pragma unroll
    for (int j = 0; j < 8; ++j) { a[j] = lo[j]; a[8 + j] = hi[j]; }
    return a;
}

// A fragment with only K=0..15 valid (attention HD=16), upper half zeroed.
__device__ __forceinline__ v16bf load_a_k16_bf(const __bf16* p, int lda,
                                               int row0, int lane) {
    int m  = lane & 15;
    int kh = (lane >> 4) << 3;
    v8bf lo = *(const v8bf*)(p + (size_t)(row0 + m) * lda + kh);
    v16bf a;
#pragma unroll
    for (int j = 0; j < 8; ++j) { a[j] = lo[j]; a[8 + j] = (__bf16)0.f; }
    return a;
}

// B^T fragment, bf16 source: two 16-byte vector loads per lane (K=kh.., 16+kh..).
__device__ __forceinline__ v16bf load_bt32_bf(const __bf16* p, int ldw, int n0,
                                              int k0, int lane) {
    int n  = lane & 15;
    int kh = (lane >> 4) << 3;
    const __bf16* r = p + (size_t)(n0 + n) * ldw + k0 + kh;
    v8bf lo = *(const v8bf*)(r);
    v8bf hi = *(const v8bf*)(r + 16);
    v16bf b;
#pragma unroll
    for (int j = 0; j < 8; ++j) { b[j] = lo[j]; b[8 + j] = hi[j]; }
    return b;
}

// B row-major gather (v matrix), bf16 source.
__device__ __forceinline__ v16bf load_b32_bf(const __bf16* p, int ldb, int k0,
                                             int n0, int lane) {
    int n  = lane & 15;
    int kh = (lane >> 4) << 3;
    const __bf16* c = p + (size_t)(k0 + kh) * ldb + n0 + n;
    v16bf b;
#pragma unroll
    for (int j = 0; j < 16; ++j) b[j] = c[(size_t)(j < 8 ? j : j + 8) * ldb];
    return b;
}

__device__ __forceinline__ v8f wmma_bf16(v16bf a, v16bf b, v8f c) {
    return __builtin_amdgcn_wmma_f32_16x16x32_bf16(false, a, false, b, (short)0, c,
                                                   false, false);
}

// ---------------------------------------------------------------------------
__global__ void zero_kernel(float* __restrict__ p, int n) {
    int i = blockIdx.x * blockDim.x + threadIdx.x;
    if (i < n) p[i] = 0.f;
}

// ---------------------------------------------------------------------------
// Input embedding: x_tile = tanh(y_seg @ W1^T + b1) @ W2^T + b2   (one group)
__global__ void embed_kernel(const float* __restrict__ y, int group, int feat,
                             const float* __restrict__ w1,  // (128, feat)
                             const float* __restrict__ b1,  // (128)
                             const float* __restrict__ w2,  // (64, 128)
                             const float* __restrict__ b2,  // (64)
                             float* __restrict__ x) {       // (B*SEQ, 64)
    __shared__ float hid[16][MLPH];
    int lane = threadIdx.x;
    int tile = blockIdx.x;                     // 0..255
    int b    = tile >> 4;
    int t0   = (tile & 15) << 4;
    int row0 = b * SEQ + group * NTOK + t0;    // row in y (ld=512) and x (ld=64)

    for (int nt = 0; nt < MLPH / 16; ++nt) {
        v8f acc = {};
        for (int k0 = 0; k0 < feat; k0 += 32) {
            __builtin_prefetch(w1 + (size_t)(nt * 16 + (lane & 15)) * feat + k0 + 32, 0, 0);
            v16bf a  = load_a32(y, MAXF, row0, k0, lane);
            v16bf bb = load_bt32(w1, feat, nt * 16, k0, lane);
            acc = wmma_bf16(a, bb, acc);
        }
        int n = lane & 15, mo = (lane >> 4) << 3;
        float bs = b1[nt * 16 + n];
#pragma unroll
        for (int r = 0; r < 8; ++r) hid[mo + r][nt * 16 + n] = fast_tanh(acc[r] + bs);
    }
    __syncthreads();
    for (int nt = 0; nt < HID / 16; ++nt) {
        v8f acc = {};
        for (int k0 = 0; k0 < MLPH; k0 += 32) {
            v16bf a  = load_a32(&hid[0][0], MLPH, 0, k0, lane);
            v16bf bb = load_bt32(w2, MLPH, nt * 16, k0, lane);
            acc = wmma_bf16(a, bb, acc);
        }
        int n = lane & 15, mo = (lane >> 4) << 3;
        float bs = b2[nt * 16 + n];
#pragma unroll
        for (int r = 0; r < 8; ++r)
            x[(size_t)(row0 + mo + r) * HID + nt * 16 + n] = acc[r] + bs;
    }
}

// ---------------------------------------------------------------------------
// QKV projection: qkv = x @ W^T + bias, stored bf16 (consumed only by WMMA).
__global__ void qkv_kernel(const float* __restrict__ A,       // (B*SEQ, 64) f32
                           const float* __restrict__ W,       // (192, 64) f32
                           const float* __restrict__ bias,    // (192)
                           __bf16* __restrict__ C) {          // (B*SEQ, 192) bf16
    int lane = threadIdx.x;
    int row0 = blockIdx.x << 4;
    int n0   = blockIdx.y << 4;
    v8f acc = {};
    for (int k0 = 0; k0 < HID; k0 += 32) {
        v16bf a  = load_a32(A, HID, row0, k0, lane);
        v16bf bb = load_bt32(W, HID, n0, k0, lane);
        acc = wmma_bf16(a, bb, acc);
    }
    int n = lane & 15, mo = (lane >> 4) << 3;
    float bs = bias[n0 + n];
#pragma unroll
    for (int r = 0; r < 8; ++r)
        C[(size_t)(row0 + mo + r) * (3 * HID) + n0 + n] = (__bf16)(acc[r] + bs);
}

// ---------------------------------------------------------------------------
// Attention: one wave per (batch, head, 16-query tile). scores tile in LDS.
// q/k/v are bf16 -> fragments load with zero conversion cost.
__global__ void attn_kernel(const __bf16* __restrict__ qkv,   // (B*SEQ, 192) bf16
                            __bf16* __restrict__ o) {         // (B*SEQ, 64) bf16
    __shared__ float S[16][SEQ];                              // 64 KB
    int lane = threadIdx.x;
    int blk  = blockIdx.x;
    int qt   = blk & 63;
    int h    = (blk >> 6) & 3;
    int b    = blk >> 8;
    const __bf16* base = qkv + (size_t)b * SEQ * 192;
    const __bf16* qp = base + h * HD;
    const __bf16* kp = base + HID + h * HD;
    const __bf16* vp = base + 2 * HID + h * HD;

    // S = (q @ k^T) / sqrt(16).  A fragment zero-pads K=16..31, so whatever
    // B holds in K=16..31 (neighbouring head columns) contributes 0.
    v16bf aq = load_a_k16_bf(qp, 192, qt * 16, lane);
    for (int nt = 0; nt < SEQ / 16; ++nt) {
        v8f acc = {};
        v16bf bk = load_bt32_bf(kp, 192, nt * 16, 0, lane);
        acc = wmma_bf16(aq, bk, acc);
        int n = lane & 15, mo = (lane >> 4) << 3;
#pragma unroll
        for (int r = 0; r < 8; ++r) S[mo + r][nt * 16 + n] = acc[r] * 0.25f;
    }
    __syncthreads();

    // row softmax (max-subtracted), 32 lanes cooperate per row
    for (int r = 0; r < 16; ++r) {
        float mx = -3.402823e38f;
        for (int c = lane; c < SEQ; c += 32) mx = fmaxf(mx, S[r][c]);
#pragma unroll
        for (int i = 16; i >= 1; i >>= 1) mx = fmaxf(mx, __shfl_xor(mx, i, 32));
        float sum = 0.f;
        for (int c = lane; c < SEQ; c += 32) {
            float e = __expf(S[r][c] - mx);
            S[r][c] = e;
            sum += e;
        }
#pragma unroll
        for (int i = 16; i >= 1; i >>= 1) sum += __shfl_xor(sum, i, 32);
        float inv = 1.f / sum;
        for (int c = lane; c < SEQ; c += 32) S[r][c] *= inv;
    }
    __syncthreads();

    // o = att @ v   (16 x 1024) @ (1024 x 16)
    v8f acc = {};
    for (int k0 = 0; k0 < SEQ; k0 += 32) {
        v16bf a  = load_a32(&S[0][0], SEQ, 0, k0, lane);
        v16bf bv = load_b32_bf(vp, 192, k0, 0, lane);
        acc = wmma_bf16(a, bv, acc);
    }
    int n = lane & 15, mo = (lane >> 4) << 3;
    int row0 = b * SEQ + qt * 16;
#pragma unroll
    for (int r = 0; r < 8; ++r)
        o[(size_t)(row0 + mo + r) * HID + h * HD + n] = (__bf16)acc[r];
}

// ---------------------------------------------------------------------------
// x = LayerNorm(x + o @ out_w^T + out_b)   (o is bf16)
__global__ void attn_out_ln_kernel(const __bf16* __restrict__ o,
                                   const float* __restrict__ ow,
                                   const float* __restrict__ ob,
                                   const float* __restrict__ lw,
                                   const float* __restrict__ lb,
                                   float* __restrict__ x) {
    __shared__ float t[16][HID];
    int lane = threadIdx.x;
    int row0 = blockIdx.x << 4;
    for (int nt = 0; nt < HID / 16; ++nt) {
        v8f acc = {};
        for (int k0 = 0; k0 < HID; k0 += 32) {
            v16bf a  = load_a32_bf(o, HID, row0, k0, lane);
            v16bf bb = load_bt32(ow, HID, nt * 16, k0, lane);
            acc = wmma_bf16(a, bb, acc);
        }
        int n = lane & 15, mo = (lane >> 4) << 3;
        float res[8];
#pragma unroll
        for (int r = 0; r < 8; ++r)
            res[r] = x[(size_t)(row0 + mo + r) * HID + nt * 16 + n];
        float bs = ob[nt * 16 + n];
#pragma unroll
        for (int r = 0; r < 8; ++r)
            t[mo + r][nt * 16 + n] = acc[r] + bs + res[r];
    }
    __syncthreads();
    if (lane < 16) {
        int r = lane;
        float m = 0.f;
        for (int c = 0; c < HID; ++c) m += t[r][c];
        m *= (1.f / HID);
        float v = 0.f;
        for (int c = 0; c < HID; ++c) { float d = t[r][c] - m; v += d * d; }
        v *= (1.f / HID);
        float inv = rsqrtf(v + 1e-5f);
        for (int c = 0; c < HID; ++c)
            x[(size_t)(row0 + r) * HID + c] = (t[r][c] - m) * inv * lw[c] + lb[c];
    }
}

// ---------------------------------------------------------------------------
// x = LayerNorm(x + relu(x @ w1^T + b1) @ w2^T + b2)
__global__ void ffn_ln_kernel(const float* __restrict__ w1,
                              const float* __restrict__ b1,
                              const float* __restrict__ w2,
                              const float* __restrict__ b2,
                              const float* __restrict__ lw,
                              const float* __restrict__ lb,
                              float* __restrict__ x) {
    __shared__ float hid[16][FF];
    __shared__ float t[16][HID];
    int lane = threadIdx.x;
    int row0 = blockIdx.x << 4;
    for (int nt = 0; nt < FF / 16; ++nt) {
        v8f acc = {};
        for (int k0 = 0; k0 < HID; k0 += 32) {
            v16bf a  = load_a32(x, HID, row0, k0, lane);
            v16bf bb = load_bt32(w1, HID, nt * 16, k0, lane);
            acc = wmma_bf16(a, bb, acc);
        }
        int n = lane & 15, mo = (lane >> 4) << 3;
        float bs = b1[nt * 16 + n];
#pragma unroll
        for (int r = 0; r < 8; ++r)
            hid[mo + r][nt * 16 + n] = fmaxf(acc[r] + bs, 0.f);
    }
    __syncthreads();
    for (int nt = 0; nt < HID / 16; ++nt) {
        v8f acc = {};
        for (int k0 = 0; k0 < FF; k0 += 32) {
            __builtin_prefetch(w2 + (size_t)(nt * 16 + (lane & 15)) * FF + k0 + 32, 0, 0);
            v16bf a  = load_a32(&hid[0][0], FF, 0, k0, lane);
            v16bf bb = load_bt32(w2, FF, nt * 16, k0, lane);
            acc = wmma_bf16(a, bb, acc);
        }
        int n = lane & 15, mo = (lane >> 4) << 3;
        float res[8];
#pragma unroll
        for (int r = 0; r < 8; ++r)
            res[r] = x[(size_t)(row0 + mo + r) * HID + nt * 16 + n];
        float bs = b2[nt * 16 + n];
#pragma unroll
        for (int r = 0; r < 8; ++r)
            t[mo + r][nt * 16 + n] = acc[r] + bs + res[r];
    }
    __syncthreads();
    if (lane < 16) {
        int r = lane;
        float m = 0.f;
        for (int c = 0; c < HID; ++c) m += t[r][c];
        m *= (1.f / HID);
        float v = 0.f;
        for (int c = 0; c < HID; ++c) { float d = t[r][c] - m; v += d * d; }
        v *= (1.f / HID);
        float inv = rsqrtf(v + 1e-5f);
        for (int c = 0; c < HID; ++c)
            x[(size_t)(row0 + r) * HID + c] = (t[r][c] - m) * inv * lw[c] + lb[c];
    }
}

// ---------------------------------------------------------------------------
// Output head per group: d = tanh(x @ w1^T + b1) @ w2^T + b2
// Writes valid columns AND the zero padding (no separate 32 MB zero pass).
// energy[b] += 0.5 * sum d^2
__global__ void head_kernel(const float* __restrict__ x, int group, int feat,
                            const float* __restrict__ w1,  // (128, 64)
                            const float* __restrict__ b1,  // (128)
                            const float* __restrict__ w2,  // (feat, 128)
                            const float* __restrict__ b2,  // (feat)
                            float* __restrict__ out,       // (B, SEQ, 512)
                            float* __restrict__ energy) {  // (B)
    __shared__ float hid[16][MLPH];
    int lane = threadIdx.x;
    int tile = blockIdx.x;
    int b    = tile >> 4;
    int t0   = (tile & 15) << 4;
    int row0 = b * SEQ + group * NTOK + t0;

    for (int nt = 0; nt < MLPH / 16; ++nt) {
        v8f acc = {};
        for (int k0 = 0; k0 < HID; k0 += 32) {
            v16bf a  = load_a32(x, HID, row0, k0, lane);
            v16bf bb = load_bt32(w1, HID, nt * 16, k0, lane);
            acc = wmma_bf16(a, bb, acc);
        }
        int n = lane & 15, mo = (lane >> 4) << 3;
        float bs = b1[nt * 16 + n];
#pragma unroll
        for (int r = 0; r < 8; ++r) hid[mo + r][nt * 16 + n] = fast_tanh(acc[r] + bs);
    }
    __syncthreads();

    float e = 0.f;
    int n = lane & 15, mo = (lane >> 4) << 3;
    for (int nt = 0; nt < (feat >> 4); ++nt) {
        v8f acc = {};
        for (int k0 = 0; k0 < MLPH; k0 += 32) {
            __builtin_prefetch(w2 + (size_t)(nt * 16 + n) * MLPH + k0 + 32, 0, 0);
            v16bf a  = load_a32(&hid[0][0], MLPH, 0, k0, lane);
            v16bf bb = load_bt32(w2, MLPH, nt * 16, k0, lane);
            acc = wmma_bf16(a, bb, acc);
        }
        float bs = b2[nt * 16 + n];
#pragma unroll
        for (int r = 0; r < 8; ++r) {
            float d = acc[r] + bs;
            out[(size_t)(row0 + mo + r) * MAXF + nt * 16 + n] = d;
            e += d * d;
        }
    }
    // zero the padded tail columns of this group's rows
    for (int nt = feat >> 4; nt < MAXF / 16; ++nt) {
#pragma unroll
        for (int r = 0; r < 8; ++r)
            out[(size_t)(row0 + mo + r) * MAXF + nt * 16 + n] = 0.f;
    }
#pragma unroll
    for (int i = 16; i >= 1; i >>= 1) e += __shfl_xor(e, i, 32);
    if (lane == 0) atomicAdd(energy + b, 0.5f * e);
}

// ---------------------------------------------------------------------------
extern "C" void kernel_launch(void* const* d_in, const int* in_sizes, int n_in,
                              void* d_out, int out_size, void* d_ws, size_t ws_size,
                              hipStream_t stream) {
    const float* y      = (const float*)d_in[1];
    const float* in_w1[4]  = {(const float*)d_in[3], (const float*)d_in[4],
                              (const float*)d_in[5], (const float*)d_in[6]};
    const float* out_w2[4] = {(const float*)d_in[7], (const float*)d_in[8],
                              (const float*)d_in[9], (const float*)d_in[10]};
    const float* out_b2[4] = {(const float*)d_in[11], (const float*)d_in[12],
                              (const float*)d_in[13], (const float*)d_in[14]};
    const float* in_b1  = (const float*)d_in[15];
    const float* in_w2  = (const float*)d_in[16];
    const float* in_b2  = (const float*)d_in[17];
    const float* out_w1 = (const float*)d_in[18];
    const float* out_b1 = (const float*)d_in[19];
    const float* qkv_w  = (const float*)d_in[20];   // (2, 192, 64)
    const float* qkv_b  = (const float*)d_in[21];   // (2, 192)
    const float* aow    = (const float*)d_in[22];   // (2, 64, 64)
    const float* aob    = (const float*)d_in[23];
    const float* ln1w   = (const float*)d_in[24];
    const float* ln1b   = (const float*)d_in[25];
    const float* fw1    = (const float*)d_in[26];   // (2, 256, 64)
    const float* fb1    = (const float*)d_in[27];
    const float* fw2    = (const float*)d_in[28];   // (2, 64, 256)
    const float* fb2    = (const float*)d_in[29];
    const float* ln2w   = (const float*)d_in[30];
    const float* ln2b   = (const float*)d_in[31];

    float* out    = (float*)d_out;
    float* energy = out + (size_t)BATCH * SEQ * MAXF;

    // workspace carve: x f32 (4 MB) | qkv bf16 (6.3 MB) | o bf16 (2 MB)
    float*  x    = (float*)d_ws;
    __bf16* qkvb = (__bf16*)(x + (size_t)BATCH * SEQ * HID);
    __bf16* ob   = qkvb + (size_t)BATCH * SEQ * 3 * HID;

    zero_kernel<<<1, 32, 0, stream>>>(energy, BATCH);

    static const int FEATS[4] = {64, 128, 256, 512};
    for (int g = 0; g < 4; ++g)
        embed_kernel<<<256, 32, 0, stream>>>(y, g, FEATS[g], in_w1[g],
                                             in_b1 + g * MLPH,
                                             in_w2 + (size_t)g * HID * MLPH,
                                             in_b2 + g * HID, x);

    for (int l = 0; l < 2; ++l) {
        qkv_kernel<<<dim3(BATCH * SEQ / 16, 3 * HID / 16), 32, 0, stream>>>(
            x, qkv_w + (size_t)l * 3 * HID * HID, qkv_b + (size_t)l * 3 * HID, qkvb);
        attn_kernel<<<BATCH * NHEAD * (SEQ / 16), 32, 0, stream>>>(qkvb, ob);
        attn_out_ln_kernel<<<BATCH * SEQ / 16, 32, 0, stream>>>(
            ob, aow + (size_t)l * HID * HID, aob + (size_t)l * HID,
            ln1w + (size_t)l * HID, ln1b + (size_t)l * HID, x);
        ffn_ln_kernel<<<BATCH * SEQ / 16, 32, 0, stream>>>(
            fw1 + (size_t)l * FF * HID, fb1 + (size_t)l * FF,
            fw2 + (size_t)l * HID * FF, fb2 + (size_t)l * HID,
            ln2w + (size_t)l * HID, ln2b + (size_t)l * HID, x);
    }

    for (int g = 0; g < 4; ++g)
        head_kernel<<<256, 32, 0, stream>>>(x, g, FEATS[g],
                                            out_w1 + (size_t)g * MLPH * HID,
                                            out_b1 + g * MLPH,
                                            out_w2[g], out_b2[g], out, energy);
}
